// BfMamba_11141145166079
// MI455X (gfx1250) — matrix operand order repaired and checked by reading, not compiled, not run on hardware
//
#include <hip/hip_runtime.h>

// ---------------- problem constants ----------------
#define BSZ   32
#define CH    256
#define LSEQ  784            // 28*28
#define DIN   512
#define DSTATE 16
#define DCONV 4
#define DTR   16
#define MROWS (BSZ * LSEQ)   // 25088, multiple of 16; LSEQ is a multiple of 16

typedef __attribute__((ext_vector_type(16))) __bf16 v16bf;
typedef __attribute__((ext_vector_type(8)))  __bf16 v8bf;
typedef __attribute__((ext_vector_type(8)))  float  v8f;

static __device__ __forceinline__ unsigned short f32_to_bf16(float f) {
  unsigned int u = __float_as_uint(f);
  u += 0x7FFFu + ((u >> 16) & 1u);     // round-to-nearest-even
  return (unsigned short)(u >> 16);
}

// ---------------- weight conversion ----------------
__global__ __launch_bounds__(256) void f32_to_bf16_kernel(
    const float* __restrict__ src, unsigned short* __restrict__ dst, int n) {
  int i = blockIdx.x * blockDim.x + threadIdx.x;
  if (i < n) dst[i] = f32_to_bf16(src[i]);
}

// x_proj_w is (48, 512); pad to (64, 512) with zero rows so N is WMMA-tileable.
__global__ __launch_bounds__(256) void xproj_pad_kernel(
    const float* __restrict__ src, unsigned short* __restrict__ dst) {
  int i = blockIdx.x * blockDim.x + threadIdx.x;     // over 64*512
  if (i < 64 * 512) {
    int r = i >> 9;                                  // / 512
    dst[i] = (r < 48) ? f32_to_bf16(src[i]) : (unsigned short)0;
  }
}

// ---------------- layernorm + transpose (B,C,HW) -> bf16 (B*L, C) ----------------
__global__ __launch_bounds__(256) void layernorm_kernel(
    const float* __restrict__ x, const float* __restrict__ w,
    const float* __restrict__ bb, unsigned short* __restrict__ xn) {
  __shared__ float red[256];
  const int m  = blockIdx.x;                // b*L + l
  const int b_ = m / LSEQ;
  const int l  = m - b_ * LSEQ;
  const int c  = threadIdx.x;
  float v = x[((size_t)b_ * CH + c) * LSEQ + l];
  red[c] = v;
  __syncthreads();
  for (int s = 128; s > 0; s >>= 1) { if (c < s) red[c] += red[c + s]; __syncthreads(); }
  float mu = red[0] * (1.0f / CH);
  __syncthreads();
  float d = v - mu;
  red[c] = d * d;
  __syncthreads();
  for (int s = 128; s > 0; s >>= 1) { if (c < s) red[c] += red[c + s]; __syncthreads(); }
  float rs = rsqrtf(red[0] * (1.0f / CH) + 1e-5f);
  xn[(size_t)m * CH + c] = f32_to_bf16(d * rs * w[c] + bb[c]);
}

// ---------------- bf16 WMMA GEMM:  C[M,N] = A[M,K] * W[N,K]^T  ----------------
// One wave computes a 16x64 output strip: one shared A fragment, four B
// fragments, FOUR independent accumulator chains (hides the bf16 WMMA->WMMA
// RAW hazard, ~5 slots) and 4x less A traffic. 4 waves/block -> 16x256 per
// block. A/B gathers use contiguous k (same in both, not the ISA table order); C/D the wave32 layout
// (CDNA5 ISA):
//   A 16x32 bf16 : lane m    -> halves K k0+0..15
//                  lane m+16 -> halves K k0+16..31
//   B 32x16 bf16 : lane n    -> halves K k0+0..15 ; lane n+16 -> K k0+16..31
//   C/D 16x16 f32: col = lane&15, row = vgpr + 8*(lane>>4)
// store_t==1 scatters the result as (B, N, LSEQ); M-tiles never straddle a
// batch because LSEQ % 16 == 0.
__global__ __launch_bounds__(128) void gemm_bf16_nt_kernel(
    const unsigned short* __restrict__ A, const unsigned short* __restrict__ Bw,
    float* __restrict__ C, int N, int K, int store_t) {
  const int lane = threadIdx.x & 31;
  const int wave = threadIdx.x >> 5;
  const int m0 = blockIdx.x * 16;
  const int n0 = (blockIdx.y * 4 + wave) * 64;   // 64 output columns per wave
  if (n0 >= N) return;                           // uniform per wave

  const int row = lane & 15;
  const int hi  = lane >> 4;
  const unsigned short* ap = A  + (size_t)(m0 + row) * K + hi * 16;
  const unsigned short* bp = Bw + (size_t)(n0 + row) * K + hi * 16;
  const size_t bstep = (size_t)16 * K;           // next 16-col sub-tile of B

  v8f acc0 = {}, acc1 = {}, acc2 = {}, acc3 = {};
  for (int k0 = 0; k0 < K; k0 += 32) {
    union { v16bf v; v8bf h[2]; } a;
    a.v = *(const v16bf*)(ap + k0);
    __builtin_prefetch(ap + k0 + 64, 0, 0);      // stream-ahead on A (HBM side)
    v16bf b0 = *(const v16bf*)(bp + k0);
    v16bf b1 = *(const v16bf*)(bp + bstep + k0);
    v16bf b2 = *(const v16bf*)(bp + 2 * bstep + k0);
    v16bf b3 = *(const v16bf*)(bp + 3 * bstep + k0);
    acc0 = __builtin_amdgcn_wmma_f32_16x16x32_bf16(false, a.v, false, b0, (short)0, acc0, false, false);
    acc1 = __builtin_amdgcn_wmma_f32_16x16x32_bf16(false, a.v, false, b1, (short)0, acc1, false, false);
    acc2 = __builtin_amdgcn_wmma_f32_16x16x32_bf16(false, a.v, false, b2, (short)0, acc2, false, false);
    acc3 = __builtin_amdgcn_wmma_f32_16x16x32_bf16(false, a.v, false, b3, (short)0, acc3, false, false);
  }

  const int col = lane & 15;
  v8f accs[4] = {acc0, acc1, acc2, acc3};
  if (!store_t) {
#pragma unroll
    for (int j = 0; j < 4; j++) {
      float* cp = C + (size_t)(m0 + hi * 8) * N + n0 + j * 16 + col;
#pragma unroll
      for (int r = 0; r < 8; r++) cp[(size_t)r * N] = accs[j][r];
    }
  } else {
#pragma unroll
    for (int j = 0; j < 4; j++) {
#pragma unroll
      for (int r = 0; r < 8; r++) {
        int m  = m0 + hi * 8 + r;
        int b_ = m / LSEQ;
        int l  = m - b_ * LSEQ;
        C[((size_t)b_ * N + (n0 + j * 16 + col)) * LSEQ + l] = accs[j][r];
      }
    }
  }
}

// ---------------- depthwise causal conv (K=4) + SiLU ----------------
__global__ __launch_bounds__(256) void conv_silu_kernel(
    const float* __restrict__ xz, const float* __restrict__ cw,
    const float* __restrict__ cb, float* __restrict__ xc,
    unsigned short* __restrict__ xcb) {
  const int m = blockIdx.x;
  const int l = m % LSEQ;
  for (int d = threadIdx.x; d < DIN; d += blockDim.x) {
    float acc = cb[d];
#pragma unroll
    for (int k = 0; k < DCONV; k++) {
      int lk = l + k - (DCONV - 1);
      if (lk >= 0) acc += xz[(size_t)(m + k - (DCONV - 1)) * 1024 + d] * cw[d * DCONV + k];
    }
    float s = acc / (1.0f + __expf(-acc));      // SiLU
    xc[(size_t)m * DIN + d]  = s;
    xcb[(size_t)m * DIN + d] = f32_to_bf16(s);
  }
}

// ---------------- dt = softplus(dt_r @ dtp_w^T + b)  (K=16, VALU) ----------------
__global__ __launch_bounds__(256) void dt_kernel(
    const float* __restrict__ xdbl, const float* __restrict__ dtw,
    const float* __restrict__ dtb, float* __restrict__ dt) {
  __shared__ float r[DTR];
  const int m = blockIdx.x;
  if (threadIdx.x < DTR) r[threadIdx.x] = xdbl[(size_t)m * 64 + threadIdx.x];
  __syncthreads();
  for (int d = threadIdx.x; d < DIN; d += blockDim.x) {
    float acc = dtb[d];
#pragma unroll
    for (int j = 0; j < DTR; j++) acc += r[j] * dtw[d * DTR + j];
    dt[(size_t)m * DIN + d] = (acc > 20.f) ? acc : __logf(1.0f + __expf(acc));
  }
}

// ---------------- selective scan: thread owns (b,d), 16 states in VGPRs ----------------
__global__ __launch_bounds__(128) void scan_kernel(
    const float* __restrict__ xdbl, const float* __restrict__ dt,
    const float* __restrict__ xc, const float* __restrict__ A_log,
    float* __restrict__ yraw) {
  const int b_ = blockIdx.x;                       // batch
  const int d  = blockIdx.y * 128 + threadIdx.x;   // channel
  __shared__ float BC[2 * DSTATE];                 // B then C for current step

  float Arow[DSTATE], h[DSTATE];
#pragma unroll
  for (int s = 0; s < DSTATE; s++) {
    Arow[s] = -__expf(A_log[d * DSTATE + s]);
    h[s] = 0.0f;
  }
  for (int l = 0; l < LSEQ; l++) {
    const int m = b_ * LSEQ + l;
    __syncthreads();
    if (threadIdx.x < 2 * DSTATE)
      BC[threadIdx.x] = xdbl[(size_t)m * 64 + DTR + threadIdx.x];
    __syncthreads();
    const float dtv = dt[(size_t)m * DIN + d];
    const float xv  = xc[(size_t)m * DIN + d];
    float y = 0.0f;
#pragma unroll
    for (int s = 0; s < DSTATE; s++) {
      float dA = __expf(dtv * Arow[s]);
      h[s] = h[s] * dA + dtv * BC[s] * xv;
      y += h[s] * BC[DSTATE + s];
    }
    yraw[(size_t)m * DIN + d] = y;
  }
}

// ---------------- y = (yraw + xc*D) * silu(z) -> bf16 ----------------
__global__ __launch_bounds__(256) void gate_kernel(
    const float* __restrict__ yraw, const float* __restrict__ xc,
    const float* __restrict__ xz, const float* __restrict__ Dp,
    unsigned short* __restrict__ yb) {
  const int m = blockIdx.x;
  for (int d = threadIdx.x; d < DIN; d += blockDim.x) {
    float y = yraw[(size_t)m * DIN + d] + xc[(size_t)m * DIN + d] * Dp[d];
    float z = xz[(size_t)m * 1024 + DIN + d];
    float g = z / (1.0f + __expf(-z));
    yb[(size_t)m * DIN + d] = f32_to_bf16(y * g);
  }
}

// ---------------- launch ----------------
extern "C" void kernel_launch(void* const* d_in, const int* in_sizes, int n_in,
                              void* d_out, int out_size, void* d_ws, size_t ws_size,
                              hipStream_t stream) {
  (void)in_sizes; (void)n_in; (void)out_size; (void)ws_size;
  const float* bbox   = (const float*)d_in[0];
  const float* norm_w = (const float*)d_in[1];
  const float* norm_b = (const float*)d_in[2];
  const float* in_w   = (const float*)d_in[3];
  const float* conv_w = (const float*)d_in[4];
  const float* conv_b = (const float*)d_in[5];
  const float* xp_w   = (const float*)d_in[6];
  const float* dtp_w  = (const float*)d_in[7];
  const float* dtp_b  = (const float*)d_in[8];
  const float* A_log  = (const float*)d_in[9];
  const float* D_par  = (const float*)d_in[10];
  const float* out_w  = (const float*)d_in[11];
  float* out = (float*)d_out;

  char* ws = (char*)d_ws;
  size_t off = 0;
  auto alloc = [&](size_t bytes) -> void* {
    void* p = ws + off;
    off = (off + bytes + 255) & ~(size_t)255;
    return p;
  };
  unsigned short* xn   = (unsigned short*)alloc((size_t)MROWS * CH * 2);
  float*          xz   = (float*)         alloc((size_t)MROWS * 1024 * 4);
  float*          xc   = (float*)         alloc((size_t)MROWS * DIN * 4);
  unsigned short* xcb  = (unsigned short*)alloc((size_t)MROWS * DIN * 2);
  float*          xdbl = (float*)         alloc((size_t)MROWS * 64 * 4);
  float*          dt   = (float*)         alloc((size_t)MROWS * DIN * 4);
  float*          yraw = (float*)         alloc((size_t)MROWS * DIN * 4);
  unsigned short* yb   = (unsigned short*)alloc((size_t)MROWS * DIN * 2);
  float*          xmid = (float*)         alloc((size_t)BSZ * CH * LSEQ * 4);
  unsigned short* wIN  = (unsigned short*)alloc((size_t)1024 * 256 * 2);
  unsigned short* wXP  = (unsigned short*)alloc((size_t)64 * 512 * 2);
  unsigned short* wOUT = (unsigned short*)alloc((size_t)256 * 512 * 2);

  for (int i = 0; i < 2; i++) {
    const float* xin = (i == 0) ? bbox : xmid;
    float* xout = (i == 1) ? out : xmid;

    f32_to_bf16_kernel<<<(1024 * 256) / 256, 256, 0, stream>>>(
        in_w + (size_t)i * 1024 * 256, wIN, 1024 * 256);
    xproj_pad_kernel<<<(64 * 512) / 256, 256, 0, stream>>>(
        xp_w + (size_t)i * 48 * 512, wXP);
    f32_to_bf16_kernel<<<(256 * 512) / 256, 256, 0, stream>>>(
        out_w + (size_t)i * 256 * 512, wOUT, 256 * 512);

    layernorm_kernel<<<MROWS, 256, 0, stream>>>(
        xin, norm_w + i * CH, norm_b + i * CH, xn);
    // in_proj: (25088,256) x (1024,256)^T -> xz (25088,1024); 4 waves x 64 cols
    gemm_bf16_nt_kernel<<<dim3(MROWS / 16, 1024 / 256), 128, 0, stream>>>(
        xn, wIN, xz, 1024, 256, 0);
    conv_silu_kernel<<<MROWS, 256, 0, stream>>>(
        xz, conv_w + i * DIN * DCONV, conv_b + i * DIN, xc, xcb);
    // x_proj: (25088,512) x (64,512)^T -> xdbl (25088,64); cols 48..63 are zero
    gemm_bf16_nt_kernel<<<dim3(MROWS / 16, 1), 128, 0, stream>>>(
        xcb, wXP, xdbl, 64, 512, 0);
    dt_kernel<<<MROWS, 256, 0, stream>>>(
        xdbl, dtp_w + i * DIN * DTR, dtp_b + i * DIN, dt);
    scan_kernel<<<dim3(BSZ, DIN / 128), 128, 0, stream>>>(
        xdbl, dt, xc, A_log + i * DIN * DSTATE, yraw);
    gate_kernel<<<MROWS, 256, 0, stream>>>(
        yraw, xc, xz, D_par + i * DIN, yb);
    // out_proj: (25088,512) x (256,512)^T -> (B, C, HW), transposed store
    gemm_bf16_nt_kernel<<<dim3(MROWS / 16, 1), 128, 0, stream>>>(
        yb, wOUT, xout, 256, 512, 1);
  }
}
